// GraphSage_78125455114733
// MI455X (gfx1250) — matrix operand order repaired and checked by reading, not compiled, not run on hardware
//
#include <hip/hip_runtime.h>
#include <stdint.h>

// GraphSage on MI455X (gfx1250, wave32):
//   agg  = mean_K gather(emb_table)            (bandwidth pass, L2-resident table)
//   h    = relu([x  | agg] @ W1^T)             (275 GFLOP -> bf16 WMMA, f32 accum)
//   out  = relu([h  | agg] @ W2^T)             (17 GFLOP  -> bf16 WMMA, f32 accum)
// GEMM staging uses GLOBAL_LOAD_ASYNC_TO_LDS (ASYNCcnt) so no VGPRs are burned
// on global->LDS data movement.

typedef __attribute__((ext_vector_type(16))) __bf16 v16bf;
typedef __attribute__((ext_vector_type(8)))  float  v8f;
typedef int vs4i __attribute__((vector_size(16)));   // matches builtin's pointee type

#define AS1 __attribute__((address_space(1)))
#define AS3 __attribute__((address_space(3)))

#if __has_builtin(__builtin_amdgcn_global_load_async_to_lds_b128)
#define USE_ASYNC_COPY 1
#else
#define USE_ASYNC_COPY 0
#endif

static constexpr int N_NODES = 4096;
static constexpr int DIM     = 4096;
static constexpr int OUTD    = 256;
static constexpr int KNB     = 6;
static constexpr int KTOT    = 2 * DIM;       // 8192 concat-K
static constexpr int KSTEP   = 32;            // bf16 WMMA K depth

__device__ __forceinline__ unsigned short f2bf(float f) {
  unsigned u = __builtin_bit_cast(unsigned, f);
  u += 0x7FFFu + ((u >> 16) & 1u);            // round-to-nearest-even
  return (unsigned short)(u >> 16);
}
__device__ __forceinline__ unsigned pack2(float a, float b) {
  return (unsigned)f2bf(a) | ((unsigned)f2bf(b) << 16);
}

__device__ __forceinline__ void wait_async0() {
#if USE_ASYNC_COPY
#if __has_builtin(__builtin_amdgcn_s_wait_asynccnt)
  __builtin_amdgcn_s_wait_asynccnt(0);
#else
  asm volatile("s_wait_asynccnt 0x0" ::: "memory");
#endif
#endif
}

// ---------------- f32 -> bf16 bulk converter (8 elems / thread / iter) -------
__global__ __launch_bounds__(256) void cvt_f32_bf16(const float* __restrict__ src,
                                                    unsigned short* __restrict__ dst,
                                                    long n8) {
  long i = (long)blockIdx.x * blockDim.x + threadIdx.x;
  const long stride = (long)gridDim.x * blockDim.x;
  const float4* s4 = (const float4*)src;
  uint4* d4 = (uint4*)dst;
  for (; i < n8; i += stride) {
    const float4 a = s4[2 * i];
    const float4 b = s4[2 * i + 1];
    uint4 o;
    o.x = pack2(a.x, a.y); o.y = pack2(a.z, a.w);
    o.z = pack2(b.x, b.y); o.w = pack2(b.z, b.w);
    d4[i] = o;
  }
}

// ---------------- neighbor mean aggregation, bf16 output ---------------------
__global__ __launch_bounds__(256) void aggregate_kernel(const float* __restrict__ emb,
                                                        const int* __restrict__ nidx,
                                                        unsigned short* __restrict__ agg) {
  const int n = blockIdx.x;
  const int t = threadIdx.x;
  int j[KNB];
#pragma unroll
  for (int k = 0; k < KNB; ++k) j[k] = nidx[n * KNB + k];
  const float inv = 1.0f / (float)KNB;
#pragma unroll
  for (int i = 0; i < 4; ++i) {
    const int c = t + 256 * i;                 // float4 chunk index (col = 4c)
    float4 s = make_float4(0.f, 0.f, 0.f, 0.f);
#pragma unroll
    for (int k = 0; k < KNB; ++k) {
      const float4 v = ((const float4*)(emb + (size_t)j[k] * DIM))[c];
      s.x += v.x; s.y += v.y; s.z += v.z; s.w += v.w;
    }
    uint2 o;
    o.x = pack2(s.x * inv, s.y * inv);
    o.y = pack2(s.z * inv, s.w * inv);
    ((uint2*)(agg + (size_t)n * DIM))[c] = o;
  }
}

// ---------------- tiled bf16 GEMM with concat-A and fused ReLU ---------------
// C[row, col] = relu( sum_k A[row,k] * B[col,k] ),  A = [Aleft | Aright]
// block tile 128x128, 8 waves (4 row-groups x 2 col-groups), wave tile 32x64,
// per-wave 2x4 fragments of v_wmma_f32_16x16x32_bf16,
// double-buffered LDS filled by async global->LDS DMA (zero staging VGPRs).
union FragBF { v16bf v; uint4 q[2]; };

template <bool OUT_BF16>
__global__ __launch_bounds__(256) void sage_gemm(const unsigned short* __restrict__ Aleft,
                                                 const unsigned short* __restrict__ Aright,
                                                 const unsigned short* __restrict__ Bw,
                                                 void* __restrict__ Cout, int ldC) {
#if USE_ASYNC_COPY
  __shared__ __align__(16) unsigned short As[2][128][40];   // 128 rows x 32 K (+8 pad)
  __shared__ __align__(16) unsigned short Bs[2][128][40];
#else
  __shared__ __align__(16) unsigned short As[1][128][40];
  __shared__ __align__(16) unsigned short Bs[1][128][40];
#endif

  const int tid  = threadIdx.x;
  const int lane = tid & 31;
  const int wave = tid >> 5;
  const int wm   = wave & 3;                  // row group: 32 rows each
  const int wn   = wave >> 2;                 // col group: 64 cols each
  const int tileRow = blockIdx.y * 128;
  const int tileCol = blockIdx.x * 128;

  // staging role: each thread owns 16 bf16 (= 32B) of one tile row
  const int lrow = tid >> 1;                  // 0..127
  const int lseg = (tid & 1) * 16;            // bf16 offset 0 or 16

  v8f acc[2][4];
#pragma unroll
  for (int m = 0; m < 2; ++m)
#pragma unroll
    for (int n = 0; n < 4; ++n) {
      v8f z = {0.f, 0.f, 0.f, 0.f, 0.f, 0.f, 0.f, 0.f};
      acc[m][n] = z;
    }

  const int KT = KTOT / KSTEP;                // 256 K-steps

  // ISA fragment layouts (CDNA5 ISA):
  //  A (16x32 bf16): lane<16 -> M=lane,     K {0..7, 16..23}
  //                  lane>=16 -> M=lane-16, K {8..15, 24..31}
  //  B (32x16 bf16): lane<16 -> N=lane,     K {0..7, 16..23} ; lane>=16 -> K {8..15, 24..31}
  //  C/D (16x16 f32): vgpr r: lane<16 -> M=r ; lane>=16 -> M=r+8 ; N=lane&15
  const int mhalf = (lane < 16) ? 0 : 8;
  const int bhalf = (lane < 16) ? 0 : 16;
  const int l15   = lane & 15;

  auto do_tile = [&](int buf) {
    FragBF af[2], bfr[4];
#pragma unroll
    for (int m = 0; m < 2; ++m) {
      const unsigned short* p = &As[buf][wm * 32 + m * 16 + l15][0];
      af[m].q[0] = *(const uint4*)(p + mhalf);
      af[m].q[1] = *(const uint4*)(p + mhalf + 16);
    }
#pragma unroll
    for (int n = 0; n < 4; ++n) {
      const unsigned short* p = &Bs[buf][wn * 64 + n * 16 + l15][0];
      bfr[n].q[0] = *(const uint4*)(p + mhalf);
      bfr[n].q[1] = *(const uint4*)(p + mhalf + 16);
    }
#pragma unroll
    for (int m = 0; m < 2; ++m)
#pragma unroll
      for (int n = 0; n < 4; ++n)
        acc[m][n] = __builtin_amdgcn_wmma_f32_16x16x32_bf16(
            false, af[m].v, false, bfr[n].v, (short)0, acc[m][n], false, false);
  };

#if USE_ASYNC_COPY
  // ------- async global->LDS DMA pipeline (ASYNCcnt), no staging VGPRs -------
  auto issue_tile = [&](int kt, int buf) {
    const int kg = kt * KSTEP;                // never straddles the 4096 boundary
    const unsigned short* s = (kg < DIM) ? Aleft : Aright;
    const unsigned short* ga =
        s + (size_t)(tileRow + lrow) * DIM + (kg & (DIM - 1)) + lseg;
    const unsigned short* gb =
        Bw + (size_t)(tileCol + lrow) * KTOT + (size_t)kt * KSTEP + lseg;
    AS1 vs4i* g0 = (AS1 vs4i*)(uintptr_t)ga;
    AS1 vs4i* g1 = (AS1 vs4i*)(uintptr_t)gb;
    // flat->LDS offset: low 32 bits of the flat address are the LDS byte offset
    AS3 vs4i* l0 = (AS3 vs4i*)(uint32_t)(uintptr_t)&As[buf][lrow][lseg];
    AS3 vs4i* l1 = (AS3 vs4i*)(uint32_t)(uintptr_t)&Bs[buf][lrow][lseg];
    // imm offset applies to BOTH global and LDS addresses (ISA 15.18.3)
    __builtin_amdgcn_global_load_async_to_lds_b128(g0, l0, 0, 0);
    __builtin_amdgcn_global_load_async_to_lds_b128(g0, l0, 16, 0);
    __builtin_amdgcn_global_load_async_to_lds_b128(g1, l1, 0, 0);
    __builtin_amdgcn_global_load_async_to_lds_b128(g1, l1, 16, 0);
  };

  issue_tile(0, 0);
  wait_async0();
  __syncthreads();

  for (int kt = 0; kt < KT; ++kt) {
    const int cur = kt & 1;
    const bool more = (kt + 1) < KT;
    if (more) issue_tile(kt + 1, cur ^ 1);    // DMA next tile while computing
    do_tile(cur);
    if (more) wait_async0();                  // own async writes to LDS done
    __syncthreads();                          // everyone's writes visible
  }
#else
  // ------- fallback: register-staged single-buffer LDS ----------------------
  for (int kt = 0; kt < KT; ++kt) {
    const int kg = kt * KSTEP;
    const unsigned short* s = (kg < DIM) ? Aleft : Aright;
    const uint4* pa =
        (const uint4*)(s + (size_t)(tileRow + lrow) * DIM + (kg & (DIM - 1)) + lseg);
    const uint4* pb =
        (const uint4*)(Bw + (size_t)(tileCol + lrow) * KTOT + (size_t)kt * KSTEP + lseg);
    uint4 ra0 = pa[0], ra1 = pa[1], rb0 = pb[0], rb1 = pb[1];
    __syncthreads();                          // previous tile fully consumed
    *(uint4*)&As[0][lrow][lseg]     = ra0;
    *(uint4*)&As[0][lrow][lseg + 8] = ra1;
    *(uint4*)&Bs[0][lrow][lseg]     = rb0;
    *(uint4*)&Bs[0][lrow][lseg + 8] = rb1;
    __syncthreads();
    do_tile(0);
  }
#endif

  // epilogue: ReLU + store (bf16 for layer-1 h, f32 for final out)
#pragma unroll
  for (int m = 0; m < 2; ++m) {
    const int rbase = tileRow + wm * 32 + m * 16 + mhalf;
#pragma unroll
    for (int n = 0; n < 4; ++n) {
      const int col = tileCol + wn * 64 + n * 16 + l15;
#pragma unroll
      for (int r = 0; r < 8; ++r) {
        float v = acc[m][n][r];
        v = v > 0.0f ? v : 0.0f;
        const size_t off = (size_t)(rbase + r) * (size_t)ldC + col;
        if (OUT_BF16) ((unsigned short*)Cout)[off] = f2bf(v);
        else          ((float*)Cout)[off] = v;
      }
    }
  }
}

extern "C" void kernel_launch(void* const* d_in, const int* in_sizes, int n_in,
                              void* d_out, int out_size, void* d_ws, size_t ws_size,
                              hipStream_t stream) {
  const float* x    = (const float*)d_in[0];   // [N, D] f32
  const int*   nidx = (const int*)d_in[1];     // [N, K] i32
  const float* emb  = (const float*)d_in[2];   // [D, D] f32
  const float* W1   = (const float*)d_in[3];   // [D, 2D] f32
  const float* W2   = (const float*)d_in[4];   // [OUT, 2D] f32
  (void)in_sizes; (void)n_in; (void)out_size; (void)ws_size;

  // workspace layout (bf16), total ~172 MB
  unsigned short* x_bf   = (unsigned short*)d_ws;                 // N*D
  unsigned short* agg_bf = x_bf   + (size_t)N_NODES * DIM;        // N*D
  unsigned short* h_bf   = agg_bf + (size_t)N_NODES * DIM;        // N*D
  unsigned short* w1_bf  = h_bf   + (size_t)N_NODES * DIM;        // D*2D
  unsigned short* w2_bf  = w1_bf  + (size_t)DIM * KTOT;           // OUT*2D

  cvt_f32_bf16<<<4096, 256, 0, stream>>>(x,  x_bf,  (long)N_NODES * DIM / 8);
  cvt_f32_bf16<<<8192, 256, 0, stream>>>(W1, w1_bf, (long)DIM * KTOT / 8);
  cvt_f32_bf16<<<1024, 256, 0, stream>>>(W2, w2_bf, (long)OUTD * KTOT / 8);

  aggregate_kernel<<<N_NODES, 256, 0, stream>>>(emb, nidx, agg_bf);

  // h = relu([x | agg] @ W1^T)  -> bf16 [N, D]
  sage_gemm<true><<<dim3(DIM / 128, N_NODES / 128), 256, 0, stream>>>(
      x_bf, agg_bf, w1_bf, (void*)h_bf, DIM);

  // out = relu([h | agg] @ W2^T) -> f32 [N, OUT]
  sage_gemm<false><<<dim3(OUTD / 128, N_NODES / 128), 256, 0, stream>>>(
      h_bf, agg_bf, w2_bf, d_out, OUTD);
}
